// ContextTransformer_41223096107635
// MI455X (gfx1250) — matrix operand order repaired and checked by reading, not compiled, not run on hardware
//
#include <hip/hip_runtime.h>

// ---------------------------------------------------------------------------
// ContextTransformer forward for MI455X (gfx1250, wave32, WMMA).
// All GEMMs run on v_wmma_f32_16x16x32_bf16 (f32 accumulate).
// Workspace requirement: ~121 MB.
// ---------------------------------------------------------------------------

#define B_    4
#define T_    1024
#define DM_   512
#define DF_   2048
#define HNUM_ 8
#define HD_   64
#define L_    6
#define MF_   256
#define RT_   2047      // 2T-1
#define ROWS_ 4096      // B*T
#define CH_   4         // (b,h) pairs processed per attention chunk

typedef __attribute__((ext_vector_type(16))) __bf16        v16bf;
typedef __attribute__((ext_vector_type(8)))  float         v8f;
typedef __attribute__((ext_vector_type(4)))  unsigned int  u32x4;

__device__ __forceinline__ unsigned short f2bf(float f) {
  unsigned int u = __float_as_uint(f);
  u += 0x7fffu + ((u >> 16) & 1u);        // round-to-nearest-even
  return (unsigned short)(u >> 16);
}

union Frag16 { v16bf v; u32x4 q[2]; };

// ---------------------------------------------------------------------------
// Generic batched WMMA GEMM:  C[M x N] = A[M x K] * BT[N x K]^T  (+ epilogue)
// A, BT are bf16 (row-major, BT is the pre-transposed B).
// Each wave computes a 32x64 tile: 2 A-frags x 4 B-frags -> 8 WMMAs / K-step.
// Epilogue modes:
//   0: row-major store            1: Q/K head-split (B,H,T,HD)
//   2: V transposed   (B,H,HD,T)  3: skew scatter-add into S (rel pos)
//   4: attention-output head merge into (B*T, DM)
// ---------------------------------------------------------------------------
__global__ void __launch_bounds__(256)
wmma_gemm(const unsigned short* __restrict__ A,  long long aBatch,
          const unsigned short* __restrict__ BT, long long bBatch,
          int M, int N, int K, int lda, int ldb,
          const float* __restrict__ bias,
          const float* __restrict__ alpha,
          const float* __restrict__ res,  long long resBatch,
          float*       __restrict__ outF, long long ofBatch,
          unsigned short* __restrict__ outB, long long obBatch,
          int mode, int p0)
{
  const int batch = blockIdx.y;
  A  += (size_t)batch * aBatch;
  BT += (size_t)batch * bBatch;
  const float*    resp = res  ? res  + (size_t)batch * resBatch : nullptr;
  float*          ofp  = outF ? outF + (size_t)batch * ofBatch  : nullptr;
  unsigned short* obp  = outB ? outB + (size_t)batch * obBatch  : nullptr;

  const int lane = threadIdx.x & 31;
  const int hl   = lane >> 4;         // lane half (0/1)
  const int lrow = lane & 15;
  const int waveId = (int)((blockIdx.x * (unsigned)blockDim.x + threadIdx.x) >> 5);
  const int tilesN = (N + 63) >> 6;
  const int tM = waveId / tilesN;
  const int tN = waveId % tilesN;
  const int m0 = tM << 5;
  const int n0 = tN << 6;
  if (m0 >= M) return;

  v8f acc[2][4];
#pragma unroll
  for (int i = 0; i < 2; ++i)
#pragma unroll
    for (int j = 0; j < 4; ++j)
#pragma unroll
      for (int v = 0; v < 8; ++v) acc[i][j][v] = 0.0f;

  // clamped row pointers (handles M=2047 / N=2047 edges)
  const unsigned short* arow[2];
#pragma unroll
  for (int i = 0; i < 2; ++i) {
    int m = m0 + i * 16 + lrow; if (m > M - 1) m = M - 1;
    arow[i] = A + (size_t)m * lda;
  }
  const unsigned short* brow[4];
#pragma unroll
  for (int j = 0; j < 4; ++j) {
    int n = n0 + j * 16 + lrow; if (n > N - 1) n = N - 1;
    brow[j] = BT + (size_t)n * ldb;
  }

  for (int ko = 0; ko < K; ko += 32) {
    Frag16 a[2], b[4];
#pragma unroll
    for (int i = 0; i < 2; ++i) {
      // A 16x32 bf16 layout: half 0 -> K {0..7,16..23}, half 1 -> K {8..15,24..31}
      a[i].q[0] = *(const u32x4*)(arow[i] + ko + hl * 8);
      a[i].q[1] = *(const u32x4*)(arow[i] + ko + 16 + hl * 8);
    }
#pragma unroll
    for (int j = 0; j < 4; ++j) {
      // B 32x16 bf16 layout: half 0 -> K {0..7,16..23}, half 1 -> K {8..15,24..31}
      const unsigned short* p = brow[j] + ko + hl * 8;
      b[j].q[0] = *(const u32x4*)(p);
      b[j].q[1] = *(const u32x4*)(p + 16);
      __builtin_prefetch(p + 64, 0, 1);   // global_prefetch_b8, two K-steps ahead
    }
#pragma unroll
    for (int i = 0; i < 2; ++i)
#pragma unroll
      for (int j = 0; j < 4; ++j)
        acc[i][j] = __builtin_amdgcn_wmma_f32_16x16x32_bf16(
            false, a[i].v, false, b[j].v, (short)0, acc[i][j], false, false);
  }

  const float aval = alpha ? alpha[0] : 0.0f;
#pragma unroll
  for (int i = 0; i < 2; ++i)
#pragma unroll
    for (int j = 0; j < 4; ++j)
#pragma unroll
      for (int v = 0; v < 8; ++v) {
        const int m = m0 + i * 16 + hl * 8 + v;   // C layout: m = 8*half + vgpr
        const int n = n0 + j * 16 + lrow;         //           n = lane & 15
        if (m >= M || n >= N) continue;
        float val = acc[i][j][v];
        if (bias)  val += bias[n];
        if (alpha) val = (val >= 0.0f) ? val : aval * val;
        size_t idx;
        if (mode == 0) {
          idx = (size_t)m * N + n;
        } else if (mode == 1) {                   // Q/K -> (B,H,T,HD)
          const int bb = m >> 10, t = m & (T_ - 1);
          const int hh = n >> 6,  d = n & 63;
          idx = (((size_t)bb * HNUM_ + hh) * T_ + t) * HD_ + d;
        } else if (mode == 2) {                   // V -> (B,H,HD,T) transposed
          const int bb = m >> 10, t = m & (T_ - 1);
          const int hh = n >> 6,  d = n & 63;
          idx = (((size_t)bb * HNUM_ + hh) * HD_ + d) * T_ + t;
        } else if (mode == 3) {                   // skew: S[q, r-(T-1)+q] += val
          const int kk = n - (T_ - 1) + m;
          if (kk >= 0 && kk < T_) ofp[(size_t)m * T_ + kk] += val;
          continue;
        } else {                                  // mode 4: O merge (B*T, DM)
          const int bh = p0 + batch;
          const int bb = bh >> 3, hh = bh & 7;
          idx = ((size_t)(bb * T_ + m)) * DM_ + hh * HD_ + n;
        }
        if (resp) val += resp[idx];
        if (ofp)  ofp[idx] = val;
        if (obp)  obp[idx] = f2bf(val);
      }
}

// ---------------------------------------------------------------------------
// LayerNorm over DM=512, one 256-thread block per row; writes bf16.
// ---------------------------------------------------------------------------
__global__ void __launch_bounds__(256)
layernorm_kernel(const float* __restrict__ x, const float* __restrict__ g,
                 const float* __restrict__ bb, unsigned short* __restrict__ out)
{
  const int row = blockIdx.x;
  const int tid = threadIdx.x;
  const float* xr = x + (size_t)row * DM_;
  __shared__ float s1[256], s2[256];
  const float v0 = xr[tid], v1 = xr[tid + 256];
  s1[tid] = v0 + v1;
  s2[tid] = v0 * v0 + v1 * v1;
  __syncthreads();
  for (int s = 128; s > 0; s >>= 1) {
    if (tid < s) { s1[tid] += s1[tid + s]; s2[tid] += s2[tid + s]; }
    __syncthreads();
  }
  const float mean = s1[0] * (1.0f / DM_);
  const float var  = s2[0] * (1.0f / DM_) - mean * mean;
  const float inv  = rsqrtf(var + 1e-5f);
  out[(size_t)row * DM_ + tid]       = f2bf((v0 - mean) * inv * g[tid] + bb[tid]);
  out[(size_t)row * DM_ + tid + 256] = f2bf((v1 - mean) * inv * g[tid + 256] + bb[tid + 256]);
}

// ---------------------------------------------------------------------------
// Softmax over k (length T) with scale + key mask; writes bf16 P.
// Grid: (T, CH) — blockIdx.y indexes the (b,h) chunk slot.
// ---------------------------------------------------------------------------
__global__ void __launch_bounds__(256)
softmax_kernel(const float* __restrict__ S, const float* __restrict__ maskv,
               unsigned short* __restrict__ P, int bh0)
{
  const int q = blockIdx.x, batch = blockIdx.y, tid = threadIdx.x;
  const int bh = bh0 + batch, bb = bh >> 3;
  const float* srow = S + ((size_t)batch * T_ + q) * T_;
  unsigned short* prow = P + ((size_t)batch * T_ + q) * T_;
  const float* mrow = maskv + (size_t)bb * T_;
  __shared__ float sm[256];

  float v[4]; float mx = -3.0e38f;
#pragma unroll
  for (int i = 0; i < 4; ++i) {
    const int k = tid + i * 256;
    v[i] = srow[k] * 0.125f + mrow[k];   // scale = HD^-0.5 = 1/8
    mx = fmaxf(mx, v[i]);
  }
  sm[tid] = mx; __syncthreads();
  for (int s = 128; s > 0; s >>= 1) { if (tid < s) sm[tid] = fmaxf(sm[tid], sm[tid + s]); __syncthreads(); }
  mx = sm[0]; __syncthreads();

  float sum = 0.0f;
#pragma unroll
  for (int i = 0; i < 4; ++i) { v[i] = __expf(v[i] - mx); sum += v[i]; }
  sm[tid] = sum; __syncthreads();
  for (int s = 128; s > 0; s >>= 1) { if (tid < s) sm[tid] += sm[tid + s]; __syncthreads(); }
  const float inv = 1.0f / sm[0];
#pragma unroll
  for (int i = 0; i < 4; ++i) prow[tid + i * 256] = f2bf(v[i] * inv);
}

// ---------------------------------------------------------------------------
// Small elementwise kernels
// ---------------------------------------------------------------------------
__global__ void maskv_kernel(const float* __restrict__ mi, float* __restrict__ maskv) {
  const int i = blockIdx.x * 256 + threadIdx.x;
  if (i >= ROWS_) return;
  const float* r = mi + (size_t)i * DM_;
  float s = 0.0f;
  for (int d = 0; d < DM_; ++d) s += r[d];
  maskv[i] = (s > 0.0f) ? 0.0f : -1.0e9f;
}

__global__ void concat_kernel(const float* __restrict__ x, const float* __restrict__ mi,
                              unsigned short* __restrict__ xm) {
  const int i = blockIdx.x * 256 + threadIdx.x;       // ROWS_*1024
  if (i >= ROWS_ * 1024) return;
  const int row = i >> 10, c = i & 1023;
  const float v = (c < 512) ? x[(size_t)row * 512 + c] : mi[(size_t)row * 512 + (c - 512)];
  xm[i] = f2bf(v);
}

__global__ void pe1_kernel(const float* __restrict__ p_kf, const float* __restrict__ w1,
                           const float* __restrict__ b1, const float* __restrict__ a,
                           unsigned short* __restrict__ out) {
  const int i = blockIdx.x * 256 + threadIdx.x;       // ROWS_*DM_
  if (i >= ROWS_ * DM_) return;
  const int row = i >> 9, n = i & 511;
  const float p0 = p_kf[(size_t)row * 2], p1 = p_kf[(size_t)row * 2 + 1];
  float v = p0 * w1[n] + p1 * w1[DM_ + n] + b1[n];
  v = (v >= 0.0f) ? v : a[0] * v;
  out[i] = f2bf(v);
}

__global__ void e1_kernel(const float* __restrict__ w1, const float* __restrict__ b1,
                          const float* __restrict__ a, unsigned short* __restrict__ out) {
  const int i = blockIdx.x * 256 + threadIdx.x;       // RT_*DM_
  if (i >= RT_ * DM_) return;
  const int row = i / DM_, n = i % DM_;
  const float r = (float)(row - (T_ - 1));
  float v = r * w1[n] + b1[n];
  v = (v >= 0.0f) ? v : a[0] * v;
  out[i] = f2bf(v);
}

__global__ void convertT_kernel(const float* __restrict__ W, unsigned short* __restrict__ WT,
                                int K, int N) {
  const int i = blockIdx.x * 256 + threadIdx.x;
  if (i >= K * N) return;
  const int k = i / N, n = i % N;
  WT[(size_t)n * K + k] = f2bf(W[i]);                 // store transposed (N x K)
}

__global__ void cvt_kernel(const float* __restrict__ in, unsigned short* __restrict__ out, int n) {
  const int i = blockIdx.x * 256 + threadIdx.x;
  if (i < n) out[i] = f2bf(in[i]);
}

// ---------------------------------------------------------------------------
// Host orchestration
// ---------------------------------------------------------------------------
extern "C" void kernel_launch(void* const* d_in, const int* in_sizes, int n_in,
                              void* d_out, int out_size, void* d_ws, size_t ws_size,
                              hipStream_t stream) {
  (void)in_sizes; (void)n_in; (void)out_size; (void)ws_size;

  const float* x        = (const float*)d_in[0];
  const float* mask_in  = (const float*)d_in[1];
  const float* p_kf     = (const float*)d_in[2];
  const float* enc_w1   = (const float*)d_in[3];
  const float* enc_b1   = (const float*)d_in[4];
  const float* enc_a1   = (const float*)d_in[5];
  const float* enc_w2   = (const float*)d_in[6];
  const float* enc_b2   = (const float*)d_in[7];
  const float* enc_a2   = (const float*)d_in[8];
  const float* kf_w1    = (const float*)d_in[9];
  const float* kf_b1    = (const float*)d_in[10];
  const float* kf_a     = (const float*)d_in[11];
  const float* kf_w2    = (const float*)d_in[12];
  const float* kf_b2    = (const float*)d_in[13];
  const float* rel_w1   = (const float*)d_in[14];
  const float* rel_b1   = (const float*)d_in[15];
  const float* rel_a    = (const float*)d_in[16];
  const float* rel_w2   = (const float*)d_in[17];
  const float* rel_b2   = (const float*)d_in[18];
  const float* ln_g     = (const float*)d_in[19];
  const float* ln_b     = (const float*)d_in[20];
  const float* attn_wq  = (const float*)d_in[21];
  const float* attn_wk  = (const float*)d_in[22];
  const float* attn_wv  = (const float*)d_in[23];
  const float* attn_wo  = (const float*)d_in[24];
  const float* attn_wob = (const float*)d_in[25];
  const float* ffn_w1   = (const float*)d_in[26];
  const float* ffn_b1   = (const float*)d_in[27];
  const float* ffn_a    = (const float*)d_in[28];
  const float* ffn_w2   = (const float*)d_in[29];
  const float* ffn_b2   = (const float*)d_in[30];
  const float* dec_w1   = (const float*)d_in[31];
  const float* dec_b1   = (const float*)d_in[32];
  const float* dec_a    = (const float*)d_in[33];
  const float* dec_w2   = (const float*)d_in[34];
  const float* dec_b2   = (const float*)d_in[35];

  char* ws = (char*)d_ws;
  size_t off = 0;
  auto alloc = [&](size_t bytes) -> void* {
    void* p = ws + off;
    off += (bytes + 255) & ~(size_t)255;
    return p;
  };

  // --- transposed bf16 weights -------------------------------------------
  unsigned short* wEnc1T = (unsigned short*)alloc((size_t)2 * 1024 * 512);
  unsigned short* wEnc2T = (unsigned short*)alloc((size_t)2 * 512 * 512);
  unsigned short* wKf2T  = (unsigned short*)alloc((size_t)2 * 512 * 512);
  unsigned short* wRel2T = (unsigned short*)alloc((size_t)2 * 512 * 64);
  unsigned short* wDec1T = (unsigned short*)alloc((size_t)2 * 512 * 512);
  unsigned short* wDec2T = (unsigned short*)alloc((size_t)2 * 512 * 256);
  unsigned short* wQT    = (unsigned short*)alloc((size_t)2 * L_ * 512 * 512);
  unsigned short* wKT    = (unsigned short*)alloc((size_t)2 * L_ * 512 * 512);
  unsigned short* wVT    = (unsigned short*)alloc((size_t)2 * L_ * 512 * 512);
  unsigned short* wOT    = (unsigned short*)alloc((size_t)2 * L_ * 512 * 512);
  unsigned short* wF1T   = (unsigned short*)alloc((size_t)2 * L_ * 512 * 2048);
  unsigned short* wF2T   = (unsigned short*)alloc((size_t)2 * L_ * 2048 * 512);

  // --- activations --------------------------------------------------------
  unsigned short* xm  = (unsigned short*)alloc((size_t)2 * ROWS_ * 1024);
  float*          h   = (float*)         alloc((size_t)4 * ROWS_ * DM_);
  unsigned short* hn  = (unsigned short*)alloc((size_t)2 * ROWS_ * DM_);
  unsigned short* Ebf = (unsigned short*)alloc((size_t)2 * RT_ * HD_);
  unsigned short* Qb  = (unsigned short*)alloc((size_t)2 * B_ * HNUM_ * T_ * HD_);
  unsigned short* Kb  = (unsigned short*)alloc((size_t)2 * B_ * HNUM_ * T_ * HD_);
  unsigned short* VTb = (unsigned short*)alloc((size_t)2 * B_ * HNUM_ * HD_ * T_);
  unsigned short* Ob  = (unsigned short*)alloc((size_t)2 * ROWS_ * DM_);
  unsigned short* t1  = (unsigned short*)alloc((size_t)2 * ROWS_ * DF_);
  float*          S   = (float*)         alloc((size_t)4 * CH_ * T_ * T_);
  unsigned short* P   = (unsigned short*)alloc((size_t)2 * CH_ * T_ * T_);
  float*          mkv = (float*)         alloc((size_t)4 * ROWS_);

  auto gemm = [&](const unsigned short* A, long long aB,
                  const unsigned short* BT, long long bB,
                  int M, int N, int K, int lda, int ldb,
                  const float* bias, const float* alpha,
                  const float* res, long long resB,
                  float* oF, long long ofB, unsigned short* oB, long long obB,
                  int mode, int p0, int batches) {
    const long long waves = (long long)((M + 31) / 32) * ((N + 63) / 64);
    const unsigned gx = (unsigned)((waves * 32 + 255) / 256);
    wmma_gemm<<<dim3(gx, batches), 256, 0, stream>>>(
        A, aB, BT, bB, M, N, K, lda, ldb, bias, alpha, res, resB,
        oF, ofB, oB, obB, mode, p0);
  };
  auto convT = [&](const float* W, unsigned short* WT, int K, int N) {
    const int tot = K * N;
    convertT_kernel<<<(tot + 255) / 256, 256, 0, stream>>>(W, WT, K, N);
  };

  // --- weight conversion (bf16, transposed) -------------------------------
  convT(enc_w1, wEnc1T, 1024, 512);
  convT(enc_w2, wEnc2T, 512, 512);
  convT(kf_w2,  wKf2T,  512, 512);
  convT(rel_w2, wRel2T, 512, 64);
  convT(dec_w1, wDec1T, 512, 512);
  convT(dec_w2, wDec2T, 512, 256);
  for (int l = 0; l < L_; ++l) {
    convT(attn_wq + (size_t)l * 512 * 512,  wQT  + (size_t)l * 512 * 512, 512, 512);
    convT(attn_wk + (size_t)l * 512 * 512,  wKT  + (size_t)l * 512 * 512, 512, 512);
    convT(attn_wv + (size_t)l * 512 * 512,  wVT  + (size_t)l * 512 * 512, 512, 512);
    convT(attn_wo + (size_t)l * 512 * 512,  wOT  + (size_t)l * 512 * 512, 512, 512);
    convT(ffn_w1  + (size_t)l * 512 * 2048, wF1T + (size_t)l * 512 * 2048, 512, 2048);
    convT(ffn_w2  + (size_t)l * 2048 * 512, wF2T + (size_t)l * 2048 * 512, 2048, 512);
  }

  // --- encoder ------------------------------------------------------------
  concat_kernel<<<(ROWS_ * 1024 + 255) / 256, 256, 0, stream>>>(x, mask_in, xm);
  maskv_kernel<<<(ROWS_ + 255) / 256, 256, 0, stream>>>(mask_in, mkv);

  // scratch reuse inside t1 (free until the layer loop's FFN)
  unsigned short* h1bf  = t1;
  unsigned short* pe1bf = t1 + (size_t)ROWS_ * DM_;
  unsigned short* e1bf  = t1 + (size_t)ROWS_ * DM_ * 2;

  // h1 = prelu(concat(x,mask) @ enc_w1 + b1, a1)
  gemm(xm, 0, wEnc1T, 0, ROWS_, 512, 1024, 1024, 1024,
       enc_b1, enc_a1, nullptr, 0, nullptr, 0, h1bf, 0, 0, 0, 1);
  // pe_kf stage 1 (K=2 -> VALU), stage 2 (512x512 -> WMMA) into f32 h
  pe1_kernel<<<(ROWS_ * DM_ + 255) / 256, 256, 0, stream>>>(p_kf, kf_w1, kf_b1, kf_a, pe1bf);
  gemm(pe1bf, 0, wKf2T, 0, ROWS_, 512, 512, 512, 512,
       kf_b2, nullptr, nullptr, 0, h, 0, nullptr, 0, 0, 0, 1);
  // h = prelu(h1 @ enc_w2 + b2, a2) + pe_kf   (residual = h, in-place safe)
  gemm(h1bf, 0, wEnc2T, 0, ROWS_, 512, 512, 512, 512,
       enc_b2, enc_a2, h, 0, h, 0, nullptr, 0, 0, 0, 1);
  // E table: E1 (VALU, K=1) then (2047 x 64) = E1 @ rel_w2 (WMMA)
  e1_kernel<<<(RT_ * DM_ + 255) / 256, 256, 0, stream>>>(rel_w1, rel_b1, rel_a, e1bf);
  gemm(e1bf, 0, wRel2T, 0, RT_, 64, 512, 512, 512,
       rel_b2, nullptr, nullptr, 0, nullptr, 0, Ebf, 0, 0, 0, 1);

  // --- transformer layers -------------------------------------------------
  for (int l = 0; l < L_; ++l) {
    const unsigned short* wq = wQT  + (size_t)l * 512 * 512;
    const unsigned short* wk = wKT  + (size_t)l * 512 * 512;
    const unsigned short* wv = wVT  + (size_t)l * 512 * 512;
    const unsigned short* wo = wOT  + (size_t)l * 512 * 512;
    const unsigned short* w1 = wF1T + (size_t)l * 512 * 2048;
    const unsigned short* w2 = wF2T + (size_t)l * 2048 * 512;

    layernorm_kernel<<<ROWS_, 256, 0, stream>>>(h, ln_g, ln_b, hn);
    gemm(hn, 0, wq, 0, ROWS_, 512, 512, 512, 512,
         nullptr, nullptr, nullptr, 0, nullptr, 0, Qb, 0, 1, 0, 1);
    gemm(hn, 0, wk, 0, ROWS_, 512, 512, 512, 512,
         nullptr, nullptr, nullptr, 0, nullptr, 0, Kb, 0, 1, 0, 1);
    gemm(hn, 0, wv, 0, ROWS_, 512, 512, 512, 512,
         nullptr, nullptr, nullptr, 0, nullptr, 0, VTb, 0, 2, 0, 1);

    for (int bh0 = 0; bh0 < B_ * HNUM_; bh0 += CH_) {
      // S = Q @ K^T   (per (b,h): T x T, K=64)
      gemm(Qb + (size_t)bh0 * T_ * HD_, (long long)T_ * HD_,
           Kb + (size_t)bh0 * T_ * HD_, (long long)T_ * HD_,
           T_, T_, HD_, HD_, HD_,
           nullptr, nullptr, nullptr, 0, S, (long long)T_ * T_, nullptr, 0, 0, 0, CH_);
      // S += skew(Q @ E^T)  (N = 2T-1, scatter-add epilogue)
      gemm(Qb + (size_t)bh0 * T_ * HD_, (long long)T_ * HD_,
           Ebf, 0,
           T_, RT_, HD_, HD_, HD_,
           nullptr, nullptr, nullptr, 0, S, (long long)T_ * T_, nullptr, 0, 3, 0, CH_);
      // P = softmax(S * scale + mask)
      softmax_kernel<<<dim3(T_, CH_), 256, 0, stream>>>(S, mkv, P, bh0);
      // O = P @ V   (A = P bf16, BT = V^T contiguous)
      gemm(P, (long long)T_ * T_,
           VTb + (size_t)bh0 * HD_ * T_, (long long)HD_ * T_,
           T_, HD_, T_, T_, T_,
           nullptr, nullptr, nullptr, 0, nullptr, 0, Ob, 0, 4, bh0, CH_);
    }

    // h = h + O @ Wo + b
    gemm(Ob, 0, wo, 0, ROWS_, 512, 512, 512, 512,
         attn_wob + (size_t)l * 512, nullptr, h, 0, h, 0, nullptr, 0, 0, 0, 1);
    // FFN
    layernorm_kernel<<<ROWS_, 256, 0, stream>>>(h, ln_g, ln_b, hn);
    gemm(hn, 0, w1, 0, ROWS_, 2048, 512, 512, 512,
         ffn_b1 + (size_t)l * 2048, ffn_a + l, nullptr, 0, nullptr, 0, t1, 0, 0, 0, 1);
    gemm(t1, 0, w2, 0, ROWS_, 512, 2048, 2048, 2048,
         ffn_b2 + (size_t)l * 512, nullptr, h, 0, h, 0, nullptr, 0, 0, 0, 1);
  }

  // --- decoder ------------------------------------------------------------
  cvt_kernel<<<(ROWS_ * DM_ + 255) / 256, 256, 0, stream>>>(h, hn, ROWS_ * DM_);
  gemm(hn, 0, wDec1T, 0, ROWS_, 512, 512, 512, 512,
       dec_b1, dec_a, nullptr, 0, nullptr, 0, Qb, 0, 0, 0, 1);   // reuse Qb
  gemm(Qb, 0, wDec2T, 0, ROWS_, 256, 512, 512, 512,
       dec_b2, nullptr, nullptr, 0, (float*)d_out, 0, nullptr, 0, 0, 0, 1);
}
